// Model_BasicNCA3D_Memory_51075751084201
// MI455X (gfx1250) — matrix operand order repaired and checked by reading, not compiled, not run on hardware
//
#include <hip/hip_runtime.h>
#include <hip/hip_bf16.h>
#include <stdint.h>

// ---------------- model constants ----------------
#define CCH   16          // visible channels
#define MEMCH 16          // memory channels
#define STCH  32          // state channels
#define HID   128
#define NSTEP 10
#define NVOX  (2*48*48*48)      // 221184
#define KCONV 432               // 27 taps * 16 ch
#define KCPAD 448               // padded to multiple of 32

// f16 weight blob layout (contiguous in workspace AND in LDS):
//   [0      , 14336) : conv  [16][448]
//   [14336  , 30720) : fc0   [128][64]
//   [30720  , 38912) : fc1   [32][128]
#define WBLOB_BYTES 38912
#define W0_OFF_H    7168        // half-element offsets
#define W1_OFF_H    15360

typedef __attribute__((ext_vector_type(16))) _Float16 v16h;
typedef __attribute__((ext_vector_type(8)))  _Float16 v8h;
typedef __attribute__((ext_vector_type(8)))  float    v8f;

static __device__ __forceinline__ v8f wmma_f16(v16h a, v16h b, v8f c) {
  // D = A(16x32 f16) * B(32x16 f16) + C(16x16 f32)
  return __builtin_amdgcn_wmma_f32_16x16x32_f16(
      /*neg_a=*/false, a, /*neg_b=*/false, b,
      /*c_mod=*/(short)0, c, /*reuse_a=*/false, /*reuse_b=*/false);
}

static __device__ __forceinline__ int refl48(int p) {
  // jnp.pad mode='reflect', pad=1: -1 -> 1, 48 -> 46
  p = (p < 0) ? -p : p;
  p = (p > 47) ? 94 - p : p;
  return p;
}

// async 16B copy global -> LDS (GLOBAL_LOAD_ASYNC_TO_LDS_B128, tracked by ASYNCcnt)
static __device__ __forceinline__ void async_copy16(const void* g, void* lds_generic) {
  // low 32 bits of a generic shared pointer are the LDS byte address (ISA 10.2)
  uint32_t loff = (uint32_t)(uintptr_t)lds_generic;
  asm volatile("global_load_async_to_lds_b128 %0, %1, off"
               :: "v"(loff), "v"(g) : "memory");
}
static __device__ __forceinline__ void async_wait0() {
  asm volatile("s_wait_asynccnt 0x0" ::: "memory");
}

// ---------------- threefry2x32 (JAX-style PRNG for the fire mask) ----------
static __device__ __forceinline__ void threefry2x32(uint32_t k0, uint32_t k1,
                                                    uint32_t x0, uint32_t x1,
                                                    uint32_t& o0, uint32_t& o1) {
  uint32_t key[3] = {k0, k1, k0 ^ k1 ^ 0x1BD11BDAu};
  const int R[8] = {13, 15, 26, 6, 17, 29, 16, 24};
  x0 += key[0]; x1 += key[1];
#pragma unroll
  for (int blk = 0; blk < 5; ++blk) {
#pragma unroll
    for (int r = 0; r < 4; ++r) {
      int rot = R[(blk & 1) * 4 + r];
      x0 += x1;
      x1 = (x1 << rot) | (x1 >> (32 - rot));
      x1 ^= x0;
    }
    x0 += key[(blk + 1) % 3];
    x1 += key[(blk + 2) % 3] + (uint32_t)(blk + 1);
  }
  o0 = x0; o1 = x1;
}

static __device__ __forceinline__ float fire_mask(int step, int v) {
  uint32_t k0, k1, o0, o1;
  threefry2x32(0u, 42u, 0u, (uint32_t)step, k0, k1);   // fold_in(key(42), step)
  const int half = NVOX / 2;
  uint32_t x0 = (v < half) ? (uint32_t)v : (uint32_t)(v - half);
  uint32_t x1 = (v < half) ? (uint32_t)(v + half) : (uint32_t)v;
  threefry2x32(k0, k1, x0, x1, o0, o1);
  uint32_t bits = (v < half) ? o0 : o1;
  float u = __uint_as_float((bits >> 9) | 0x3f800000u) - 1.0f; // [0,1)
  return (u > 0.5f) ? 1.0f : 0.0f;                              // FIRE_RATE
}

// ---------------- weight pre-conversion: fp32 -> f16, [N][K] layout --------
__global__ void prep_weights(const float* __restrict__ conv_w,
                             const float* __restrict__ fc0_w,
                             const float* __restrict__ fc1_w,
                             _Float16* __restrict__ wBlob) {
  int t = blockIdx.x * 256 + threadIdx.x;
  if (t < 16 * KCPAD) {                         // conv [16][448]
    int o = t / KCPAD, k = t % KCPAD;
    float val = 0.0f;
    if (k < KCONV) {
      int i = k & 15, tap = k >> 4;             // k = tap*16 + in_ch
      val = conv_w[o * 432 + i * 27 + tap];     // conv_w[o][i][a][b][c]
    }
    wBlob[t] = (_Float16)val;
  }
  if (t < 128 * 64) {                           // fc0 [128][64], K pad 48->64
    int n = t / 64, k = t % 64;
    wBlob[W0_OFF_H + t] = (_Float16)((k < 48) ? fc0_w[n * 48 + k] : 0.0f);
  }
  if (t < 32 * 128) {                           // fc1 [32][128]
    wBlob[W1_OFF_H + t] = (_Float16)fc1_w[t];
  }
}

// ---------------- state init: state[...,:16]=x, [...,16:]=0 ---------------
__global__ void init_state(const float* __restrict__ x, float* __restrict__ s) {
  int t = blockIdx.x * 256 + threadIdx.x;
  if (t >= NVOX * STCH) return;
  int v = t >> 5, c = t & 31;
  s[t] = (c < CCH) ? x[v * CCH + c] : 0.0f;
}

// ---------------- one NCA step: conv + MLP fully on WMMA ------------------
__launch_bounds__(128)
__global__ void nca_step(const float* __restrict__ src, float* __restrict__ dst,
                         const _Float16* __restrict__ wBlob,
                         const float* __restrict__ conv_b,
                         const float* __restrict__ fc0_b,
                         int step) {
  __shared__ _Float16 wL[WBLOB_BYTES / 2];   // all weights, f16, async-loaded
  __shared__ _Float16 convD[4][16 * 16];     // per-wave conv-result transpose
  __shared__ _Float16 hA[4][16 * HID];       // per-wave hidden activations
  __shared__ float    maskSh[4][16];

  const int lane = threadIdx.x & 31;
  const int wv   = threadIdx.x >> 5;
  const int lh   = lane & 15;                // row (A) / column (B,D)
  const int hi   = lane >> 4;                // half-wave selector
  const int hi8  = hi * 8;

  // ---- async weight staging: 38,912 B = 2432 x 16 B, all 128 lanes ----
  {
    const char* gb = (const char*)wBlob;
    char*       lb = (char*)wL;
    for (int i = threadIdx.x; i < WBLOB_BYTES / 16; i += 128)
      async_copy16(gb + i * 16, lb + i * 16);
  }

  const int tile  = blockIdx.x * 4 + wv;     // NVOX/16 tiles exactly
  const int vbase = tile * 16;

  // decode this lane's A-row voxel (vbase + lh)
  const int v  = vbase + lh;
  int w  = v % 48;
  int tq = v / 48;
  int h  = tq % 48; tq /= 48;
  int d  = tq % 48;
  const int bb = tq / 48;

  // reflected neighbor coordinates (computed once)
  int dzc[3] = {refl48(d - 1), d, refl48(d + 1)};
  int hyc[3] = {refl48(h - 1), h, refl48(h + 1)};
  int wxc[3] = {refl48(w - 1), w, refl48(w + 1)};

  if (hi == 0) maskSh[wv][lh] = fire_mask(step, v);

  async_wait0();                 // ASYNCcnt==0: weights resident in LDS
  __syncthreads();

  const _Float16* wcL = wL;                   // [16][448]
  const _Float16* w0L = wL + W0_OFF_H;        // [128][64]
  const _Float16* w1L = wL + W1_OFF_H;        // [32][128]

  // neighbor pointer for a tap: tap = a*9 + b*3 + c -> (h+a-1, w+b-1, d+c-1)
  auto nbr = [&](int tap) -> const float* {
    int aa = tap / 9, rr = tap % 9, bx = rr / 3, cz = rr % 3;
    return src + ((size_t)(((bb * 48 + dzc[cz]) * 48 + hyc[aa]) * 48 + wxc[bx]) * STCH + hi8);
  };

  // ---------------- conv as GEMM: D(16x16) = A(16x448) * B(448x16) + bias --
  v8f cacc;
#pragma unroll
  for (int r = 0; r < 8; ++r) cacc[r] = conv_b[lh];     // C init = bias[N]

#pragma unroll
  for (int kc = 0; kc < 14; ++kc) {                     // fully unrolled: taps const
    v16h a;
    {   // halves 0..7: tap 2*kc, channels i = hi8 + j
      const float* p = nbr(2 * kc);
      float4 u0 = *(const float4*)p;
      float4 u1 = *(const float4*)(p + 4);
      a[0] = (_Float16)u0.x; a[1] = (_Float16)u0.y; a[2] = (_Float16)u0.z; a[3] = (_Float16)u0.w;
      a[4] = (_Float16)u1.x; a[5] = (_Float16)u1.y; a[6] = (_Float16)u1.z; a[7] = (_Float16)u1.w;
    }
    if (2 * kc + 1 < 27) {   // halves 8..15: tap 2*kc+1 (compile-time test)
      const float* p = nbr(2 * kc + 1);
      float4 u0 = *(const float4*)p;
      float4 u1 = *(const float4*)(p + 4);
      a[8]  = (_Float16)u0.x; a[9]  = (_Float16)u0.y; a[10] = (_Float16)u0.z; a[11] = (_Float16)u0.w;
      a[12] = (_Float16)u1.x; a[13] = (_Float16)u1.y; a[14] = (_Float16)u1.z; a[15] = (_Float16)u1.w;
    } else {
#pragma unroll
      for (int j = 8; j < 16; ++j) a[j] = (_Float16)0.0f;   // K pad 432->448
    }
    v16h b = __builtin_shufflevector(*(const v8h*)(wcL + lh * KCPAD + kc * 32 + hi8), *(const v8h*)(wcL + lh * KCPAD + kc * 32 + 16 + hi8), 0, 1, 2, 3, 4, 5, 6, 7, 8, 9, 10, 11, 12, 13, 14, 15);
    cacc = wmma_f16(a, b, cacc);
  }

  // stage conv result for the A-side transpose: convD[M][N]
#pragma unroll
  for (int r = 0; r < 8; ++r)
    convD[wv][(hi8 + r) * 16 + lh] = (_Float16)cacc[r];
  __syncthreads();

  // ---------------- fc0: D(16x128) = A(16x64) * B(64x128) + b, ReLU --------
  v16h a0, a1;
  {
    const float* p = src + (size_t)(v * STCH + hi8);          // xc channels
    float4 u0 = *(const float4*)p;
    float4 u1 = *(const float4*)(p + 4);
    a0[0] = (_Float16)u0.x; a0[1] = (_Float16)u0.y; a0[2] = (_Float16)u0.z; a0[3] = (_Float16)u0.w;
    a0[4] = (_Float16)u1.x; a0[5] = (_Float16)u1.y; a0[6] = (_Float16)u1.z; a0[7] = (_Float16)u1.w;
    // conv channels: contiguous 8 halfs -> single ds_load_b128
    v8h cv = *(const v8h*)(convD[wv] + lh * 16 + hi8);
#pragma unroll
    for (int j = 0; j < 8; ++j) a0[8 + j] = cv[j];
    const float* q = src + (size_t)(v * STCH + 16 + hi8);     // xm channels
    float4 m0 = *(const float4*)q;
    float4 m1 = *(const float4*)(q + 4);
    a1[0] = (_Float16)m0.x; a1[1] = (_Float16)m0.y; a1[2] = (_Float16)m0.z; a1[3] = (_Float16)m0.w;
    a1[4] = (_Float16)m1.x; a1[5] = (_Float16)m1.y; a1[6] = (_Float16)m1.z; a1[7] = (_Float16)m1.w;
#pragma unroll
    for (int j = 8; j < 16; ++j) a1[j] = (_Float16)0.0f;      // K pad 48->64
  }

#pragma unroll
  for (int nt = 0; nt < 8; ++nt) {
    v8f hacc;
#pragma unroll
    for (int r = 0; r < 8; ++r) hacc[r] = fc0_b[nt * 16 + lh];
    v16h b0 = __builtin_shufflevector(*(const v8h*)(w0L + (nt * 16 + lh) * 64 + 0 + hi8), *(const v8h*)(w0L + (nt * 16 + lh) * 64 + 16 + hi8), 0, 1, 2, 3, 4, 5, 6, 7, 8, 9, 10, 11, 12, 13, 14, 15);
    hacc = wmma_f16(a0, b0, hacc);
    v16h b1 = __builtin_shufflevector(*(const v8h*)(w0L + (nt * 16 + lh) * 64 + 32 + hi8), *(const v8h*)(w0L + (nt * 16 + lh) * 64 + 48 + hi8), 0, 1, 2, 3, 4, 5, 6, 7, 8, 9, 10, 11, 12, 13, 14, 15);
    hacc = wmma_f16(a1, b1, hacc);
#pragma unroll
    for (int r = 0; r < 8; ++r) {
      float rv = fmaxf(hacc[r], 0.0f);                        // ReLU
      hA[wv][(hi8 + r) * HID + nt * 16 + lh] = (_Float16)rv;
    }
  }
  __syncthreads();

  // ---------------- fc1: D(16x32) = A(16x128) * B(128x32), masked update ---
  v16h fa[4];
#pragma unroll
  for (int kc = 0; kc < 4; ++kc) {
    // two contiguous 8-half runs -> two ds_load_b128 (vs 16 scalar u16 loads)
    v8h lo = *(const v8h*)(hA[wv] + lh * HID + kc * 32 + hi8);
    v8h hi2 = *(const v8h*)(hA[wv] + lh * HID + kc * 32 + 16 + hi8);
#pragma unroll
    for (int j = 0; j < 8; ++j) {
      fa[kc][j]     = lo[j];
      fa[kc][8 + j] = hi2[j];
    }
  }
#pragma unroll
  for (int nt = 0; nt < 2; ++nt) {
    v8f dacc;
#pragma unroll
    for (int r = 0; r < 8; ++r) dacc[r] = 0.0f;
#pragma unroll
    for (int kc = 0; kc < 4; ++kc) {
      v16h b = __builtin_shufflevector(*(const v8h*)(w1L + (nt * 16 + lh) * HID + kc * 32 + hi8), *(const v8h*)(w1L + (nt * 16 + lh) * HID + kc * 32 + 16 + hi8), 0, 1, 2, 3, 4, 5, 6, 7, 8, 9, 10, 11, 12, 13, 14, 15);
      dacc = wmma_f16(fa[kc], b, dacc);
    }
#pragma unroll
    for (int r = 0; r < 8; ++r) {
      int M  = hi8 + r;
      int vv = vbase + M;
      int ch = nt * 16 + lh;
      float old = src[(size_t)vv * STCH + ch];
      float nv  = (ch == 0) ? old : old + dacc[r] * maskSh[wv][M];
      dst[(size_t)vv * STCH + ch] = nv;
    }
  }
}

// ---------------- host-side orchestration --------------------------------
extern "C" void kernel_launch(void* const* d_in, const int* in_sizes, int n_in,
                              void* d_out, int out_size, void* d_ws, size_t ws_size,
                              hipStream_t stream) {
  const float* x      = (const float*)d_in[0];
  const float* conv_w = (const float*)d_in[1];
  const float* conv_b = (const float*)d_in[2];
  const float* fc0_w  = (const float*)d_in[3];
  const float* fc0_b  = (const float*)d_in[4];
  const float* fc1_w  = (const float*)d_in[5];

  char* ws = (char*)d_ws;
  float*    stateWs = (float*)ws;                              // NVOX*32 f32
  size_t off = (size_t)NVOX * STCH * sizeof(float);            // 28,311,552 B (256-aligned)
  _Float16* wBlob = (_Float16*)(ws + off);                     // 38,912 B f16 blob

  float* outState = (float*)d_out;

  prep_weights<<<32, 256, 0, stream>>>(conv_w, fc0_w, fc1_w, wBlob);
  init_state<<<(NVOX * STCH + 255) / 256, 256, 0, stream>>>(x, outState);

  // ping-pong: even steps d_out -> ws, odd steps ws -> d_out; step 9 ends in d_out
  for (int s = 0; s < NSTEP; ++s) {
    const float* src = (s & 1) ? stateWs  : outState;
    float*       dst = (s & 1) ? outState : stateWs;
    nca_step<<<NVOX / 64, 128, 0, stream>>>(src, dst, wBlob, conv_b, fc0_b, s);
  }
}
